// Qwen2_5_VlVisionTower_876173328769
// MI455X (gfx1250) — matrix operand order repaired and checked by reading, not compiled, not run on hardware
//
#include <hip/hip_runtime.h>
#include <math.h>

// ---- problem constants ----
#define S_TOK 2304
#define DM    1280
#define NH    16
#define HD    80
#define HDP   96      // HD padded to multiple of 32 for WMMA K
#define FF    3420
#define FFP   3424    // FF padded to multiple of 32
#define NL    4
#define IND   1176
#define INDP  1184    // padded to multiple of 32
#define OUTD  3584
#define MD    5120
#define SM    576
#define WTOK  64
#define EPSV  1e-6f

typedef _Float16 v16h __attribute__((ext_vector_type(16)));
typedef _Float16 v8h  __attribute__((ext_vector_type(8)));
typedef float    v8f  __attribute__((ext_vector_type(8)));

union HF { v16h v16; v8h h[2]; };

__device__ inline v8h zero8h() {
  v8h z;
#pragma unroll
  for (int i = 0; i < 8; ++i) z[i] = (_Float16)0.f;
  return z;
}

// A fragment: 16x32 f16, row-major source with leading dim ld (in elements).
// lane<16: row=lane,    K {0..7} in h[0], K {16..23} in h[1]
// lane>=16: row=lane-16, K {8..15} in h[0], K {24..31} in h[1]
__device__ inline v16h load_frag_a(const _Float16* base, int ld, int lane) {
  const int row = lane & 15;
  const int k0  = (lane & 16) ? 8 : 0;
  const _Float16* p = base + (long)row * ld + k0;
  HF f;
  f.h[0] = *(const v8h*)p;
  f.h[1] = *(const v8h*)(p + 16);
  return f.v16;
}

// B fragment: 32x16 f16 = W[n][k] (W row-major N x K, ld elements).
// lane<16: col=lane, K {0..7},{16..23} ; lane>=16: col=lane-16, K {8..15},{24..31}
__device__ inline v16h load_frag_b(const _Float16* base, int ld, int lane) {
  const int row = lane & 15;
  const int k0  = (lane & 16) ? 8 : 0;
  const _Float16* p = base + (long)row * ld + k0;
  HF f;
  f.h[0] = *(const v8h*)p;
  f.h[1] = *(const v8h*)(p + 16);
  return f.v16;
}

// ---- fp32 -> f16 with zero padding (N x K -> Np x Kp) ----
__global__ void cvt_f16_pad(const float* __restrict__ src, _Float16* __restrict__ dst,
                            int N, int K, int Kp, long total) {
  long i = (long)blockIdx.x * 256 + threadIdx.x;
  if (i >= total) return;
  int n = (int)(i / Kp), k = (int)(i % Kp);
  float v = (n < N && k < K) ? src[(long)n * K + k] : 0.f;
  dst[i] = (_Float16)v;
}

// ---- generic WMMA GEMM: C(M x N) = A16(M x Kp) * W16(Np x Kp)^T (+bias, epilogues) ----
// one wave per 32 x (16*NT) tile; NT is compile-time so the K-loop has no branches.
// mode: 0 store f32, 1 += f32 (residual), 2 gelu->f16, 3 scatter-row f32
template <int NT>
__global__ void __launch_bounds__(32, 1)
gemm_f16(const _Float16* __restrict__ A, const _Float16* __restrict__ W,
         const float* __restrict__ bias, float* __restrict__ outF,
         _Float16* __restrict__ outH, const int* __restrict__ scat,
         int n_off, int Nreal, int Kp, int ldo, int mode) {
  const int lane = threadIdx.x & 31;
  const int m0 = blockIdx.x * 32;
  const int n0 = n_off + blockIdx.y * (16 * NT);
  const _Float16* a0 = A + (long)m0 * Kp;
  const _Float16* a1 = a0 + (long)16 * Kp;
  const _Float16* wb = W + (long)n0 * Kp;

  v8f acc[2][NT];
#pragma unroll
  for (int mt = 0; mt < 2; ++mt)
#pragma unroll
    for (int t = 0; t < NT; ++t)
#pragma unroll
      for (int i = 0; i < 8; ++i) acc[mt][t][i] = 0.f;

  const int kit = Kp >> 5;
  for (int kk = 0; kk < kit; ++kk) {
    v16h af0 = load_frag_a(a0 + kk * 32, Kp, lane);
    v16h af1 = load_frag_a(a1 + kk * 32, Kp, lane);
    __builtin_prefetch(a0 + kk * 32 + 1024, 0, 1);
#pragma unroll
    for (int t = 0; t < NT; ++t) {
      v16h bf = load_frag_b(wb + (long)t * 16 * Kp + kk * 32, Kp, lane);
      acc[0][t] = __builtin_amdgcn_wmma_f32_16x16x32_f16(false, af0, false, bf,
                                                         (short)0, acc[0][t], false, false);
      acc[1][t] = __builtin_amdgcn_wmma_f32_16x16x32_f16(false, af1, false, bf,
                                                         (short)0, acc[1][t], false, false);
    }
  }

  const int nl = lane & 15;
  const int mh = (lane & 16) ? 8 : 0;
#pragma unroll
  for (int mt = 0; mt < 2; ++mt)
#pragma unroll
    for (int t = 0; t < NT; ++t) {
      int n = n0 + t * 16 + nl;
      float b = (bias != nullptr && n < Nreal) ? bias[n] : 0.f;
#pragma unroll
      for (int r = 0; r < 8; ++r) {
        int m = m0 + mt * 16 + mh + r;
        float v = acc[mt][t][r] + b;
        if (mode == 0)      outF[(long)m * ldo + n] = v;
        else if (mode == 1) outF[(long)m * ldo + n] += v;
        else if (mode == 2) outH[(long)m * ldo + n] =
            (_Float16)(0.5f * v * (1.f + erff(v * 0.70710678118654752f)));
        else                outF[(long)scat[m] * ldo + n] = v;
      }
    }
}

// ---- RMSNorm (per-token) -> f16 ----
__global__ void rmsnorm_f16(const float* __restrict__ x, const float* __restrict__ g,
                            _Float16* __restrict__ outp) {
  __shared__ float red[8];
  const int row = blockIdx.x, tid = threadIdx.x;
  const float* xr = x + (long)row * DM;
  float ss = 0.f;
  for (int d = tid; d < DM; d += 256) { float v = xr[d]; ss += v * v; }
  for (int off = 16; off > 0; off >>= 1) ss += __shfl_xor(ss, off, 32);
  if ((tid & 31) == 0) red[tid >> 5] = ss;
  __syncthreads();
  if (tid == 0) { float t = 0.f; for (int i = 0; i < 8; ++i) t += red[i]; red[0] = t; }
  __syncthreads();
  const float sc = rsqrtf(red[0] / (float)DM + EPSV);
  for (int d = tid; d < DM; d += 256)
    outp[(long)row * DM + d] = (_Float16)(g[d] * xr[d] * sc);
}

// ---- window reorder of patch-embed rows ----
__global__ void reorder_rows(const float* __restrict__ src, float* __restrict__ dst,
                             const int* __restrict__ wi) {
  long i = (long)blockIdx.x * 256 + threadIdx.x;
  if (i >= (long)S_TOK * DM) return;
  int s = (int)(i / DM), d = (int)(i % DM);
  int srow = wi[s >> 2] * 4 + (s & 3);
  dst[i] = src[(long)srow * DM + d];
}

// ---- cos/sin tables in window order: emb = concat(rpe,rpe) ----
__global__ void build_rot(const float* __restrict__ rpe, const int* __restrict__ wi,
                          float* __restrict__ cosb, float* __restrict__ sinb) {
  int i = blockIdx.x * 256 + threadIdx.x;
  if (i >= S_TOK * HD) return;
  int s = i / HD, j = i % HD;
  int src = wi[s >> 2] * 4 + (s & 3);
  float a = rpe[src * (HD / 2) + (j % (HD / 2))];
  cosb[i] = cosf(a);
  sinb[i] = sinf(a);
}

// ---- QKV (S x 3840 f32) -> rope'd q16/k16 (H,S,HDP) + transposed v (H,HD,S) ----
__global__ void rope_pack(const float* __restrict__ qkv, const float* __restrict__ cosb,
                          const float* __restrict__ sinb, _Float16* __restrict__ q16,
                          _Float16* __restrict__ k16, _Float16* __restrict__ vt16) {
  int i = blockIdx.x * 256 + threadIdx.x;
  if (i >= S_TOK * NH * HDP) return;
  int j = i % HDP;
  int sh = i / HDP;
  int h = sh % NH;
  int s = sh / NH;
  long qidx = ((long)h * S_TOK + s) * HDP + j;
  if (j >= HD) { q16[qidx] = (_Float16)0.f; k16[qidx] = (_Float16)0.f; return; }
  long base = (long)s * (3 * DM) + h * HD + j;
  float c  = cosb[s * HD + j];
  float sn = sinb[s * HD + j];
  int jr = (j < HD / 2) ? (j + HD / 2) : (j - HD / 2);
  long rb = (long)s * (3 * DM) + h * HD + jr;
  float sgn = (j < HD / 2) ? -1.f : 1.f;
  float q = qkv[base];
  float k = qkv[base + DM];
  float v = qkv[base + 2 * DM];
  float qr = sgn * qkv[rb];
  float kr = sgn * qkv[rb + DM];
  q16[qidx] = (_Float16)(q * c + qr * sn);
  k16[qidx] = (_Float16)(k * c + kr * sn);
  vt16[((long)h * HD + j) * S_TOK + s] = (_Float16)v;
}

// ---- flash attention: one wave per (16-query tile, head) ----
// Incremental per-lane pointers: no per-iteration 64-bit address math.
__global__ void __launch_bounds__(32, 1)
attn_kernel(const _Float16* __restrict__ q16,
            const _Float16* __restrict__ k16,
            const _Float16* __restrict__ vt16,
            _Float16* __restrict__ o16, int full) {
  __shared__ _Float16 lp[256];
  const int lane = threadIdx.x & 31;
  const int q0 = blockIdx.x * 16;
  const int h  = blockIdx.y;
  const int kstart = full ? 0 : (q0 / WTOK) * WTOK;
  const int ntiles = (full ? S_TOK : WTOK) / 16;
  const float scl = 0.11180339887498948f;  // 1/sqrt(80)

  const _Float16* qb = q16 + ((long)h * S_TOK + q0) * HDP;
  v16h aq0 = load_frag_a(qb, HDP, lane);
  v16h aq1 = load_frag_a(qb + 32, HDP, lane);
  v16h aq2 = load_frag_a(qb + 64, HDP, lane);

  const int mh = (lane & 16) ? 8 : 0;
  const int nl = lane & 15;

  // per-lane K pointer: row = key (lane&15), chunk offsets j*32 (+16) are immediates
  const _Float16* kl = k16 + ((long)h * S_TOK + kstart + nl) * HDP + ((lane & 16) ? 8 : 0);
  // per-lane V^T pointer (both half-waves real; upper one adds 8 keys at load): row = hd column, + key offset
  const _Float16* vl = vt16 + ((long)h * HD + nl) * S_TOK + kstart;
  // per-lane LDS addresses for the P transpose
  _Float16* lps = &lp[(mh /*row*/) * 16 + nl];            // store: row mh+r, col nl
  const _Float16* lpl = &lp[nl * 16 + ((lane & 16) ? 8 : 0)];  // load: row nl, k0

  float mrow[8], lrow[8];
  v8f o[5];
#pragma unroll
  for (int r = 0; r < 8; ++r) { mrow[r] = -3.0e38f; lrow[r] = 0.f; }
#pragma unroll
  for (int j = 0; j < 5; ++j)
#pragma unroll
    for (int r = 0; r < 8; ++r) o[j][r] = 0.f;

  for (int kt = 0; kt < ntiles; ++kt) {
    // ---- S = Q K^T over padded HD=96 ----
    HF b0, b1, b2;
    b0.h[0] = *(const v8h*)(kl + 0);   b0.h[1] = *(const v8h*)(kl + 16);
    b1.h[0] = *(const v8h*)(kl + 32);  b1.h[1] = *(const v8h*)(kl + 48);
    b2.h[0] = *(const v8h*)(kl + 64);  b2.h[1] = *(const v8h*)(kl + 80);
    v8f s;
#pragma unroll
    for (int r = 0; r < 8; ++r) s[r] = 0.f;
    s = __builtin_amdgcn_wmma_f32_16x16x32_f16(false, aq0, false, b0.v16, (short)0, s, false, false);
    s = __builtin_amdgcn_wmma_f32_16x16x32_f16(false, aq1, false, b1.v16, (short)0, s, false, false);
    s = __builtin_amdgcn_wmma_f32_16x16x32_f16(false, aq2, false, b2.v16, (short)0, s, false, false);

    // ---- online softmax; rows of C live across a 16-lane half-wave ----
#pragma unroll
    for (int r = 0; r < 8; ++r) {
      float v = s[r] * scl;
      float mx = v;
      for (int off = 1; off < 16; off <<= 1) mx = fmaxf(mx, __shfl_xor(mx, off, 32));
      float mnew = fmaxf(mrow[r], mx);
      float fac = __expf(mrow[r] - mnew);
      float pe  = __expf(v - mnew);
      float sm = pe;
      for (int off = 1; off < 16; off <<= 1) sm += __shfl_xor(sm, off, 32);
      lrow[r] = lrow[r] * fac + sm;
      mrow[r] = mnew;
#pragma unroll
      for (int j = 0; j < 5; ++j) o[j][r] *= fac;
      lps[r * 16] = (_Float16)pe;  // C-layout -> (m,n) in LDS
    }
    asm volatile("s_wait_dscnt 0x0" ::: "memory");

    // reload P as an A fragment (K = 16 real keys, upper 16 zero)
    HF pf;
    pf.h[0] = *(const v8h*)lpl;
    pf.h[1] = zero8h();
    asm volatile("s_wait_dscnt 0x0" ::: "memory");

    // ---- O += P V  (V^T rows contiguous in keys; h[0] = keys 8h..8h+7, h[1] is the zero K half) ----
#pragma unroll
    for (int j = 0; j < 5; ++j) {
      HF bf;
      bf.h[0] = *(const v8h*)(vl + (long)j * 16 * S_TOK + ((lane & 16) ? 8 : 0));
      bf.h[1] = zero8h();
      o[j] = __builtin_amdgcn_wmma_f32_16x16x32_f16(false, pf.v16, false, bf.v16,
                                                    (short)0, o[j], false, false);
    }
    kl += 16 * HDP;  // next 16 keys
    vl += 16;
  }

#pragma unroll
  for (int j = 0; j < 5; ++j)
#pragma unroll
    for (int r = 0; r < 8; ++r) {
      int m = q0 + mh + r;
      o16[(long)m * DM + h * HD + j * 16 + nl] = (_Float16)(o[j][r] / lrow[r]);
    }
}

// ---- silu(gate)*up -> f16 ----
__global__ void silu_mul(const float* __restrict__ g, const float* __restrict__ u,
                         _Float16* __restrict__ outp, long n) {
  long i = (long)blockIdx.x * 256 + threadIdx.x;
  if (i >= n) return;
  float gv = g[i];
  float sg = gv / (1.f + __expf(-gv));
  outp[i] = (_Float16)(sg * u[i]);
}

extern "C" void kernel_launch(void* const* d_in, const int* in_sizes, int n_in,
                              void* d_out, int out_size, void* d_ws, size_t ws_size,
                              hipStream_t stream) {
  (void)in_sizes; (void)n_in; (void)out_size; (void)ws_size;
  const float* hs   = (const float*)d_in[0];
  const float* rpe  = (const float*)d_in[1];
  const float* wpat = (const float*)d_in[2];
  const float* g1   = (const float*)d_in[3];
  const float* wqkv = (const float*)d_in[4];
  const float* bqkv = (const float*)d_in[5];
  const float* wo   = (const float*)d_in[6];
  const float* bo   = (const float*)d_in[7];
  const float* g2   = (const float*)d_in[8];
  const float* wg   = (const float*)d_in[9];
  const float* bg   = (const float*)d_in[10];
  const float* wu   = (const float*)d_in[11];
  const float* bu   = (const float*)d_in[12];
  const float* wd   = (const float*)d_in[13];
  const float* bd   = (const float*)d_in[14];
  const float* gq   = (const float*)d_in[15];
  const float* wm1  = (const float*)d_in[16];
  const float* bm1  = (const float*)d_in[17];
  const float* wm2  = (const float*)d_in[18];
  const float* bm2  = (const float*)d_in[19];
  const int*   wi   = (const int*)d_in[20];
  float* out = (float*)d_out;

  char* ws = (char*)d_ws;
  size_t off = 0;
  auto take = [&](size_t bytes) {
    char* p = ws + off;
    off = (off + bytes + 255) & ~(size_t)255;
    return p;
  };
  float*    xf32 = (float*)take((size_t)S_TOK * DM * 4);
  _Float16* h16  = (_Float16*)take((size_t)S_TOK * DM * 2);   // also 576x5120 merger A
  _Float16* ff16 = (_Float16*)take((size_t)S_TOK * FFP * 2);  // ffn A / patch A / merger mid
  float*    tmp1 = (float*)take((size_t)S_TOK * 3 * DM * 4);  // qkv / gate
  float*    tmp2 = (float*)take((size_t)S_TOK * FFP * 4);     // up / pre-reorder patch
  _Float16* w16  = (_Float16*)take((size_t)MD * MD * 2);      // current weight (max wm1)
  float*    cosb = (float*)take((size_t)S_TOK * HD * 4);
  float*    sinb = (float*)take((size_t)S_TOK * HD * 4);
  _Float16* q16  = (_Float16*)take((size_t)NH * S_TOK * HDP * 2);
  _Float16* k16  = (_Float16*)take((size_t)NH * S_TOK * HDP * 2);
  _Float16* vt16 = (_Float16*)take((size_t)NH * HD * S_TOK * 2);

  auto cvt = [&](const float* src, _Float16* dst, int N, int K, int Np, int Kp) {
    long cnt = (long)Np * Kp;
    cvt_f16_pad<<<(int)((cnt + 255) / 256), 256, 0, stream>>>(src, dst, N, K, Kp, cnt);
  };
  // exact-cover launches: full 64-wide tiles with NT=4, remainder (Np%64)/16 tiles separately
  auto gemm = [&](const _Float16* A, const _Float16* Wt, const float* bias, void* outp,
                  const int* scat, int M, int Nreal, int Np, int Kp, int ldo, int mode) {
    int nfull = Np / 64;
    int rem = (Np % 64) / 16;
    if (nfull > 0)
      gemm_f16<4><<<dim3(M / 32, nfull), 32, 0, stream>>>(
          A, Wt, bias, (float*)outp, (_Float16*)outp, scat, 0, Nreal, Kp, ldo, mode);
    int n_off = nfull * 64;
    if (rem == 1)
      gemm_f16<1><<<dim3(M / 32, 1), 32, 0, stream>>>(
          A, Wt, bias, (float*)outp, (_Float16*)outp, scat, n_off, Nreal, Kp, ldo, mode);
    else if (rem == 2)
      gemm_f16<2><<<dim3(M / 32, 1), 32, 0, stream>>>(
          A, Wt, bias, (float*)outp, (_Float16*)outp, scat, n_off, Nreal, Kp, ldo, mode);
    else if (rem == 3)
      gemm_f16<3><<<dim3(M / 32, 1), 32, 0, stream>>>(
          A, Wt, bias, (float*)outp, (_Float16*)outp, scat, n_off, Nreal, Kp, ldo, mode);
  };

  // ---- patch embed: (S x 1176) @ w_patch^T ----
  cvt(hs, ff16, S_TOK, IND, S_TOK, INDP);
  cvt(wpat, w16, DM, IND, DM, INDP);
  gemm(ff16, w16, nullptr, tmp2, nullptr, S_TOK, DM, DM, INDP, DM, 0);
  reorder_rows<<<(S_TOK * DM + 255) / 256, 256, 0, stream>>>(tmp2, xf32, wi);
  build_rot<<<(S_TOK * HD + 255) / 256, 256, 0, stream>>>(rpe, wi, cosb, sinb);

  for (int l = 0; l < NL; ++l) {
    // attention block
    rmsnorm_f16<<<S_TOK, 256, 0, stream>>>(xf32, g1 + (size_t)l * DM, h16);
    cvt(wqkv + (size_t)l * 3 * DM * DM, w16, 3 * DM, DM, 3 * DM, DM);
    gemm(h16, w16, bqkv + (size_t)l * 3 * DM, tmp1, nullptr, S_TOK, 3 * DM, 3 * DM, DM, 3 * DM, 0);
    rope_pack<<<(S_TOK * NH * HDP + 255) / 256, 256, 0, stream>>>(tmp1, cosb, sinb, q16, k16, vt16);
    {
      dim3 g(S_TOK / 16, NH);
      attn_kernel<<<g, 32, 0, stream>>>(q16, k16, vt16, h16, (l == 3) ? 1 : 0);
    }
    cvt(wo + (size_t)l * DM * DM, w16, DM, DM, DM, DM);
    gemm(h16, w16, bo + (size_t)l * DM, xf32, nullptr, S_TOK, DM, DM, DM, DM, 1);

    // MLP block
    rmsnorm_f16<<<S_TOK, 256, 0, stream>>>(xf32, g2 + (size_t)l * DM, h16);
    cvt(wg + (size_t)l * FF * DM, w16, FF, DM, FFP, DM);
    gemm(h16, w16, bg + (size_t)l * FF, tmp1, nullptr, S_TOK, FF, FFP, DM, FFP, 0);
    cvt(wu + (size_t)l * FF * DM, w16, FF, DM, FFP, DM);
    gemm(h16, w16, bu + (size_t)l * FF, tmp2, nullptr, S_TOK, FF, FFP, DM, FFP, 0);
    silu_mul<<<(int)(((long)S_TOK * FFP + 255) / 256), 256, 0, stream>>>(
        tmp1, tmp2, ff16, (long)S_TOK * FFP);
    cvt(wd + (size_t)l * DM * FF, w16, DM, FF, DM, FFP);
    gemm(ff16, w16, bd + (size_t)l * DM, xf32, nullptr, S_TOK, DM, DM, FFP, DM, 1);
  }

  // ---- merger: rms -> (576 x 5120) -> gelu MLP -> scatter-undo window order ----
  rmsnorm_f16<<<S_TOK, 256, 0, stream>>>(xf32, gq, h16);  // contiguous == (576,5120)
  cvt(wm1, w16, MD, MD, MD, MD);
  gemm(h16, w16, bm1, ff16, nullptr, SM, MD, MD, MD, MD, 2);     // gelu -> f16
  cvt(wm2, w16, OUTD, MD, OUTD, MD);
  gemm(ff16, w16, bm2, out, wi, SM, OUTD, OUTD, MD, OUTD, 3);    // scatter rows
}
